// ClusterForecasting_66769561583760
// MI455X (gfx1250) — compile-verified, matrix operand order checked by reading, not run on hardware
//
#include <hip/hip_runtime.h>
#include <math.h>

// ---------------------------------------------------------------------------
// Problem dims (from reference)
// ---------------------------------------------------------------------------
#define BB 8
#define FFEAT 8
#define DD 256
#define HH 8
#define LLAYERS 2
#define FFN 1024
#define KC 5
#define SEQ 512              // S*T
#define ROWS 4096            // B*SEQ
#define DHD 32               // D/H
#define NPTS 256             // B*S
#define ENC_IN 4096          // T*D

typedef __attribute__((ext_vector_type(16))) _Float16 v16h;
typedef __attribute__((ext_vector_type(8)))  float    v8f;
typedef __attribute__((ext_vector_type(4)))  unsigned int u32x4;
typedef __attribute__((ext_vector_type(8)))  int      i32x8;
typedef __attribute__((ext_vector_type(4)))  int      i32x4;
typedef unsigned int       u32;
typedef unsigned long long u64;

// ---------------------------------------------------------------------------
// TDM: async-load one 32x32 f32 tile (row stride = stride_elems) into LDS.
// D# layout per CDNA5 ISA 8.3/8.4: group0 {count=1, lds_addr, global_addr,
// type=2}; group1 {data_size=4B, tensor_dim0/1=32, tile_dim0/1=32,
// tensor_dim0_stride=stride_elems}.  Groups 2/3 zero (2D tensor).
// ---------------------------------------------------------------------------
__device__ __forceinline__ void tdm_load_tile32(u32 lds_off, const void* gptr,
                                                u32 stride_elems)
{
    const u64 ga = (u64)(size_t)gptr;
    u32x4 g0;
    g0[0] = 1u;                                           // count=1 (valid, user)
    g0[1] = lds_off;                                      // lds_addr (bytes)
    g0[2] = (u32)ga;                                      // global_addr[31:0]
    g0[3] = (u32)((ga >> 32) & 0x01ffffffu) | (2u << 30); // addr[56:32] | type=2

    const u64 q0 = (0x2ull << 16)                         // data_size = 4 bytes
                 | (32ull << 48);                         // tensor_dim0[15:0]
    const u64 q1 = (32ull << 16)                          // tensor_dim1 (bits 111:80)
                 | (32ull << 48);                         // tile_dim0 (bits 127:112)
    const u64 q2 = 32ull                                  // tile_dim1 (bits 143:128)
                 | ((u64)stride_elems << 32);             // tensor_dim0_stride lo32
    const u64 q3 = 0ull;
    i32x8 g1;
    g1[0] = (int)(u32)q0; g1[1] = (int)(u32)(q0 >> 32);
    g1[2] = (int)(u32)q1; g1[3] = (int)(u32)(q1 >> 32);
    g1[4] = (int)(u32)q2; g1[5] = (int)(u32)(q2 >> 32);
    g1[6] = (int)(u32)q3; g1[7] = (int)(u32)(q3 >> 32);
    i32x4 z4 = {};
#if __clang_major__ >= 23
    i32x8 z8 = {};
    __builtin_amdgcn_tensor_load_to_lds(g0, g1, z4, z4, z8, 0);
#else
    __builtin_amdgcn_tensor_load_to_lds(g0, g1, z4, z4, 0);
#endif
}

// ---------------------------------------------------------------------------
// Load one f16 A-fragment (16x32 layout, ISA 7.12.2) for this lane.
// p = &A[row][k0 + hf*8]: two contiguous 8-float runs at +0 and +16.
// Emits 4x global_load_b128.
// ---------------------------------------------------------------------------
__device__ __forceinline__ v16h load_frag_row(const float* __restrict__ p)
{
    const float4 x0 = *(const float4*)(p);
    const float4 x1 = *(const float4*)(p + 4);
    const float4 x2 = *(const float4*)(p + 16);
    const float4 x3 = *(const float4*)(p + 20);
    v16h a;
    a[0]=(_Float16)x0.x;  a[1]=(_Float16)x0.y;  a[2]=(_Float16)x0.z;  a[3]=(_Float16)x0.w;
    a[4]=(_Float16)x1.x;  a[5]=(_Float16)x1.y;  a[6]=(_Float16)x1.z;  a[7]=(_Float16)x1.w;
    a[8]=(_Float16)x2.x;  a[9]=(_Float16)x2.y;  a[10]=(_Float16)x2.z; a[11]=(_Float16)x2.w;
    a[12]=(_Float16)x3.x; a[13]=(_Float16)x3.y; a[14]=(_Float16)x3.z; a[15]=(_Float16)x3.w;
    return a;
}

__device__ __forceinline__ float epilogue(float v, float bv, int act)
{
    v += bv;
    if (act == 1)      v = fmaxf(v, 0.0f);
    else if (act == 2) v = 1.0f / (1.0f + __expf(-v));
    return v;
}

__device__ __forceinline__ void store_tile(float* __restrict__ C, int ldc,
                                           int m0, int n, int hf,
                                           const v8f& acc, float bv,
                                           float alpha, int act)
{
#pragma unroll
    for (int r = 0; r < 8; ++r) {
        const int m = m0 + r + (hf << 3);
        C[(long)m * ldc + n] = epilogue(acc[r] * alpha, bv, act);
    }
}

#define WMMA(a, b, c) __builtin_amdgcn_wmma_f32_16x16x32_f16( \
        false, (a), false, (b), (short)0, (c), false, false)

// ---------------------------------------------------------------------------
// Fast GEMM, B not transposed:  C = act(alpha * A @ B + bias)
// Requires M,N,K % 32 == 0, lda % 4 == 0.  One wave per 32x32 C tile.
// B tiles streamed into LDS by the Tensor Data Mover with ping-pong double
// buffering: tile i+1 is issued before tile i is consumed, and the wave only
// waits for TENSORcnt <= 1 (tensor ops complete in order), overlapping the
// DMA with the WMMA work.  A is loaded directly as b128.
// grid = (N/32, M/32, batch), block = 32.
// ---------------------------------------------------------------------------
__global__ void wmma_gemm_tdm(const float* __restrict__ A, int lda, long sA,
                              const float* __restrict__ B, int ldb, long sB,
                              float* __restrict__ C, int ldc, long sC,
                              const float* __restrict__ bias,
                              int M, int N, int K, float alpha, int act)
{
    __shared__ float lb[2][32 * 32];   // ping-pong B tiles [k][n], TDM-written
    const long bz = blockIdx.z;
    A += bz * sA;  B += bz * sB;  C += bz * sC;

    const int n0   = blockIdx.x * 32;
    const int m0   = blockIdx.y * 32;
    const int lane = threadIdx.x & 31;
    const int ml   = lane & 15;
    const int hf   = lane >> 4;
    const u32 lds0 = (u32)(size_t)(void*)&lb[0][0];
    const u32 lds1 = (u32)(size_t)(void*)&lb[1][0];

    const float* a0p = A + (long)(m0 + ml) * lda + (hf << 3);
    const float* a1p = A + (long)(m0 + 16 + ml) * lda + (hf << 3);
    const float* bp  = B + n0;

    v8f acc00 = {}, acc01 = {}, acc10 = {}, acc11 = {};

    // prologue: start tile 0
    tdm_load_tile32(lds0, bp, (u32)ldb);

    for (int k0 = 0; k0 < K; k0 += 32) {
        const int cur = (k0 >> 5) & 1;
        const float* lbc = lb[cur];

        const v16h a0 = load_frag_row(a0p + k0);
        const v16h a1 = load_frag_row(a1p + k0);

        if (k0 + 32 < K) {                                   // uniform branch
            tdm_load_tile32(cur ? lds0 : lds1,
                            bp + (long)(k0 + 32) * ldb, (u32)ldb);
            __builtin_prefetch(a0p + k0 + 32, 0, 3);
            __builtin_prefetch(a1p + k0 + 32, 0, 3);
            __builtin_amdgcn_s_wait_tensorcnt(1);            // current tile done
        } else {
            __builtin_amdgcn_s_wait_tensorcnt(0);
        }
        __asm__ volatile("" ::: "memory");

        v16h b0, b1;
#pragma unroll
        for (int g = 0; g < 8; ++g) {
            const int kk = ((g & 4) << 2) + (hf << 3) + ((g & 3) << 1);
            b0[2*g]   = (_Float16)lbc[kk * 32 + ml];
            b0[2*g+1] = (_Float16)lbc[(kk + 1) * 32 + ml];
            b1[2*g]   = (_Float16)lbc[kk * 32 + 16 + ml];
            b1[2*g+1] = (_Float16)lbc[(kk + 1) * 32 + 16 + ml];
        }
        acc00 = WMMA(a0, b0, acc00);
        acc01 = WMMA(a0, b1, acc01);
        acc10 = WMMA(a1, b0, acc10);
        acc11 = WMMA(a1, b1, acc11);
    }

    const int nA = n0 + ml, nB = n0 + 16 + ml;
    const float bvA = bias ? bias[nA] : 0.0f;
    const float bvB = bias ? bias[nB] : 0.0f;
    store_tile(C, ldc, m0,      nA, hf, acc00, bvA, alpha, act);
    store_tile(C, ldc, m0,      nB, hf, acc01, bvB, alpha, act);
    store_tile(C, ldc, m0 + 16, nA, hf, acc10, bvA, alpha, act);
    store_tile(C, ldc, m0 + 16, nB, hf, acc11, bvB, alpha, act);
}

// ---------------------------------------------------------------------------
// Fast GEMM, B transposed (C = alpha * A @ B^T): B rows contiguous in K, so
// both operands use vectorized b128 fragment loads.  Used for Q @ K^T.
// Requires M,N,K % 32 == 0.  grid = (N/32, M/32, batch), block = 32.
// ---------------------------------------------------------------------------
__global__ void wmma_gemm_tn(const float* __restrict__ A, int lda, long sA,
                             const float* __restrict__ B, int ldb, long sB,
                             float* __restrict__ C, int ldc, long sC,
                             int M, int N, int K, float alpha)
{
    const long bz = blockIdx.z;
    A += bz * sA;  B += bz * sB;  C += bz * sC;

    const int n0   = blockIdx.x * 32;
    const int m0   = blockIdx.y * 32;
    const int lane = threadIdx.x & 31;
    const int ml   = lane & 15;
    const int hf   = lane >> 4;

    const float* a0p = A + (long)(m0 + ml) * lda + (hf << 3);
    const float* a1p = A + (long)(m0 + 16 + ml) * lda + (hf << 3);
    const float* b0p = B + (long)(n0 + ml) * ldb + (hf << 3);
    const float* b1p = B + (long)(n0 + 16 + ml) * ldb + (hf << 3);

    v8f acc00 = {}, acc01 = {}, acc10 = {}, acc11 = {};

    for (int k0 = 0; k0 < K; k0 += 32) {
        const v16h a0 = load_frag_row(a0p + k0);
        const v16h a1 = load_frag_row(a1p + k0);
        const v16h b0 = load_frag_row(b0p + k0);
        const v16h b1 = load_frag_row(b1p + k0);
        acc00 = WMMA(a0, b0, acc00);
        acc01 = WMMA(a0, b1, acc01);
        acc10 = WMMA(a1, b0, acc10);
        acc11 = WMMA(a1, b1, acc11);
    }

    const int nA = n0 + ml, nB = n0 + 16 + ml;
    store_tile(C, ldc, m0,      nA, hf, acc00, 0.0f, alpha, 0);
    store_tile(C, ldc, m0,      nB, hf, acc01, 0.0f, alpha, 0);
    store_tile(C, ldc, m0 + 16, nA, hf, acc10, 0.0f, alpha, 0);
    store_tile(C, ldc, m0 + 16, nB, hf, acc11, 0.0f, alpha, 0);
}

// ---------------------------------------------------------------------------
// Guarded WMMA GEMM for ragged shapes (only N=2 / K=2 tiny launches).
// ---------------------------------------------------------------------------
__global__ void wmma_gemm_safe(const float* __restrict__ A, int lda,
                               const float* __restrict__ B, int ldb,
                               float* __restrict__ C, int ldc,
                               const float* __restrict__ bias,
                               int M, int N, int K, int act)
{
    const int n0   = blockIdx.x * 16;
    const int m0   = blockIdx.y * 16;
    const int lane = threadIdx.x & 31;
    const int ml   = lane & 15;
    const int hf   = lane >> 4;

    const int arow = m0 + ml;
    const int bcol = n0 + ml;
    const bool aok = (arow < M);
    const bool bok = (bcol < N);

    v8f acc = {};
    for (int k0 = 0; k0 < K; k0 += 32) {
        v16h af, bf;
#pragma unroll
        for (int g = 0; g < 8; ++g) {
            const int kk = ((g & 4) << 2) + (hf << 3) + ((g & 3) << 1);
            const int k1 = k0 + kk, k2 = k1 + 1;
            float a0 = (aok && k1 < K) ? A[(long)arow * lda + k1] : 0.0f;
            float a1 = (aok && k2 < K) ? A[(long)arow * lda + k2] : 0.0f;
            af[2*g] = (_Float16)a0;  af[2*g+1] = (_Float16)a1;
            float b0 = (bok && k1 < K) ? B[(long)k1 * ldb + bcol] : 0.0f;
            float b1 = (bok && k2 < K) ? B[(long)k2 * ldb + bcol] : 0.0f;
            bf[2*g] = (_Float16)b0;  bf[2*g+1] = (_Float16)b1;
        }
        acc = WMMA(af, bf, acc);
    }
    if (!bok) return;
    const float bv = bias ? bias[bcol] : 0.0f;
#pragma unroll
    for (int r = 0; r < 8; ++r) {
        const int m = m0 + r + (hf << 3);
        if (m < M) C[(long)m * ldc + bcol] = epilogue(acc[r], bv, act);
    }
}

// ---------------------------------------------------------------------------
// Embedding: h[r,d] = sum_f x[r,f]*W[f,d] + b[d]   (K=8 -> plain VALU)
// ---------------------------------------------------------------------------
__global__ void embed_kernel(const float* __restrict__ x,
                             const float* __restrict__ W,
                             const float* __restrict__ b,
                             float* __restrict__ h)
{
    const long i = (long)blockIdx.x * 256 + threadIdx.x;
    const long r = i >> 8;
    const int  d = (int)(i & 255);
    float acc = b[d];
#pragma unroll
    for (int f = 0; f < FFEAT; ++f)
        acc += x[r * FFEAT + f] * W[f * DD + d];
    h[i] = acc;
}

// ---------------------------------------------------------------------------
// Row softmax over `cols` entries.  grid = (rows, heads), block = 256.
// ---------------------------------------------------------------------------
__global__ void softmax_kernel(float* __restrict__ s, int cols)
{
    float* p = s + ((long)blockIdx.y * gridDim.x + blockIdx.x) * cols;
    const int t = threadIdx.x;
    __shared__ float red[256];

    float m = -1e30f;
    for (int c = t; c < cols; c += 256) m = fmaxf(m, p[c]);
    red[t] = m; __syncthreads();
    for (int o = 128; o > 0; o >>= 1) { if (t < o) red[t] = fmaxf(red[t], red[t+o]); __syncthreads(); }
    m = red[0]; __syncthreads();

    float sum = 0.0f;
    for (int c = t; c < cols; c += 256) { float e = __expf(p[c] - m); p[c] = e; sum += e; }
    red[t] = sum; __syncthreads();
    for (int o = 128; o > 0; o >>= 1) { if (t < o) red[t] += red[t+o]; __syncthreads(); }
    const float inv = 1.0f / red[0];
    for (int c = t; c < cols; c += 256) p[c] *= inv;
}

// ---------------------------------------------------------------------------
// z = LayerNorm(z + o) * g + b  over D=256.  grid = ROWS, block = 256.
// ---------------------------------------------------------------------------
__global__ void add_ln_kernel(float* __restrict__ z, const float* __restrict__ o,
                              const float* __restrict__ g, const float* __restrict__ b)
{
    const long base = (long)blockIdx.x * DD;
    const int t = threadIdx.x;
    __shared__ float red[256];

    const float v = z[base + t] + o[base + t];
    red[t] = v; __syncthreads();
    for (int s = 128; s > 0; s >>= 1) { if (t < s) red[t] += red[t+s]; __syncthreads(); }
    const float mu = red[0] * (1.0f / DD); __syncthreads();

    const float d = v - mu;
    red[t] = d * d; __syncthreads();
    for (int s = 128; s > 0; s >>= 1) { if (t < s) red[t] += red[t+s]; __syncthreads(); }
    const float var = red[0] * (1.0f / DD);

    z[base + t] = d * rsqrtf(var + 1e-5f) * g[t] + b[t];
}

// ---------------------------------------------------------------------------
// Stage-1 L1 reduction: part[blk] = sum |a-b| over 1024 elems.  1024 blocks.
// ---------------------------------------------------------------------------
__global__ void l1_partial_kernel(const float* __restrict__ a,
                                  const float* __restrict__ b,
                                  float* __restrict__ part, long n)
{
    const int t = threadIdx.x;
    const long base = ((long)blockIdx.x * 256 + t) * 4;
    float s = 0.0f;
#pragma unroll
    for (int j = 0; j < 4; ++j) {
        const long i = base + j;
        if (i < n) s += fabsf(a[i] - b[i]);
    }
    __shared__ float red[256];
    red[t] = s; __syncthreads();
    for (int o = 128; o > 0; o >>= 1) { if (t < o) red[t] += red[t+o]; __syncthreads(); }
    if (t == 0) part[blockIdx.x] = red[0];
}

// ---------------------------------------------------------------------------
// Finalize: cluster assignment, segment stats, KL, loss.  1 block x 256.
// out layout: [loss, kl, idx(256), low(512)]
// ---------------------------------------------------------------------------
__global__ void finalize_kernel(const float* __restrict__ low,
                                const float* __restrict__ centers,
                                const float* __restrict__ part, int npart,
                                float* __restrict__ out, float logN, float recDen)
{
    const int t = threadIdx.x;
    __shared__ float red[256];
    __shared__ int   s_idx[256];
    __shared__ float s_dsq[256];
    __shared__ float s_sum[KC], s_cnt[KC];
    __shared__ float s_rec, s_inter;

    const float lx = low[2*t], ly = low[2*t+1];
    float best = 3.402823e38f; int bi = 0;
#pragma unroll
    for (int c = 0; c < KC; ++c) {
        const float dx = lx - centers[2*c], dy = ly - centers[2*c+1];
        const float d = dx*dx + dy*dy;
        if (d < best) { best = d; bi = c; }
    }
    s_idx[t] = bi; s_dsq[t] = best;

    float rs = 0.0f;
    for (int j = t; j < npart; j += 256) rs += part[j];
    red[t] = rs; __syncthreads();
    for (int o = 128; o > 0; o >>= 1) { if (t < o) red[t] += red[t+o]; __syncthreads(); }
    if (t == 0) s_rec = red[0];
    __syncthreads();

    red[t] = best; __syncthreads();
    for (int o = 128; o > 0; o >>= 1) { if (t < o) red[t] += red[t+o]; __syncthreads(); }
    if (t == 0) s_inter = red[0];
    __syncthreads();

    if (t < KC) {   // deterministic serial segment sums
        float s = 0.0f, c = 0.0f;
        for (int n = 0; n < NPTS; ++n)
            if (s_idx[n] == t) { s += s_dsq[n]; c += 1.0f; }
        s_sum[t] = s; s_cnt[t] = c;
    }
    __syncthreads();

    if (t == 0) {
        const float inter = s_inter * (1.0f / NPTS);
        float intra_sum = 0.0f, ne = 0.0f;
        for (int c = 0; c < KC; ++c)
            if (s_cnt[c] > 0.0f) { intra_sum += s_sum[c] / s_cnt[c]; ne += 1.0f; }
        const float intra = intra_sum / fmaxf(ne, 1.0f);
        // arr = zeros(N).at[idx].set(100): ne entries of 100, rest 0.
        const float lse  = 100.0f + logf(ne + ((float)NPTS - ne) * __expf(-100.0f));
        const float invN = 1.0f / (float)NPTS;
        const float sum_logp = ne * (100.0f - lse) + ((float)NPTS - ne) * (-lse);
        const float kl = invN * (invN * ((float)NPTS * (-logN) - sum_logp));
        out[0] = inter + intra + kl + s_rec / recDen;
        out[1] = kl;
    }
    out[2 + t]     = (float)bi;
    out[258 + 2*t] = lx;
    out[259 + 2*t] = ly;
}

// ---------------------------------------------------------------------------
// Host side
// ---------------------------------------------------------------------------
static inline void run_tdm(const float* A, int lda, long sA,
                           const float* B, int ldb, long sB,
                           float* C, int ldc, long sC, const float* bias,
                           int M, int N, int K, float alpha, int act,
                           int batch, hipStream_t st)
{
    dim3 grid(N / 32, M / 32, batch);
    wmma_gemm_tdm<<<grid, dim3(32, 1, 1), 0, st>>>(A, lda, sA, B, ldb, sB,
                                                   C, ldc, sC, bias, M, N, K, alpha, act);
}

static inline void run_tn(const float* A, int lda, long sA,
                          const float* B, int ldb, long sB,
                          float* C, int ldc, long sC,
                          int M, int N, int K, float alpha, int batch, hipStream_t st)
{
    dim3 grid(N / 32, M / 32, batch);
    wmma_gemm_tn<<<grid, dim3(32, 1, 1), 0, st>>>(A, lda, sA, B, ldb, sB,
                                                  C, ldc, sC, M, N, K, alpha);
}

static inline void run_safe(const float* A, int lda, const float* B, int ldb,
                            float* C, int ldc, const float* bias,
                            int M, int N, int K, int act, hipStream_t st)
{
    dim3 grid((N + 15) / 16, (M + 15) / 16, 1);
    wmma_gemm_safe<<<grid, dim3(32, 1, 1), 0, st>>>(A, lda, B, ldb, C, ldc,
                                                    bias, M, N, K, act);
}

extern "C" void kernel_launch(void* const* d_in, const int* in_sizes, int n_in,
                              void* d_out, int out_size, void* d_ws, size_t ws_size,
                              hipStream_t stream)
{
    const float* x_seg  = (const float*)d_in[1];
    const float* emb_W  = (const float*)d_in[2];
    const float* emb_b  = (const float*)d_in[3];
    const float* Wq     = (const float*)d_in[4];
    const float* Wk     = (const float*)d_in[5];
    const float* Wv     = (const float*)d_in[6];
    const float* Wo     = (const float*)d_in[7];
    const float* W1     = (const float*)d_in[8];
    const float* b1     = (const float*)d_in[9];
    const float* W2     = (const float*)d_in[10];
    const float* b2     = (const float*)d_in[11];
    const float* ln1_g  = (const float*)d_in[12];
    const float* ln1_b  = (const float*)d_in[13];
    const float* ln2_g  = (const float*)d_in[14];
    const float* ln2_b  = (const float*)d_in[15];
    const float* enc_W1 = (const float*)d_in[16];
    const float* enc_b1 = (const float*)d_in[17];
    const float* enc_W2 = (const float*)d_in[18];
    const float* enc_b2 = (const float*)d_in[19];
    const float* dec_W1 = (const float*)d_in[20];
    const float* dec_b1 = (const float*)d_in[21];
    const float* dec_W2 = (const float*)d_in[22];
    const float* dec_b2 = (const float*)d_in[23];
    const float* centers= (const float*)d_in[24];
    float* out = (float*)d_out;

    // workspace layout (floats)
    float* w = (float*)d_ws;
    const size_t M1 = 1u << 20;
    float* z      = w;                    // [4096,256] == pts_in [256,4096]
    float* qb     = w + 1*M1;
    float* kb     = w + 2*M1;
    float* vb     = w + 3*M1;
    float* attb   = w + 4*M1;
    float* ob     = w + 5*M1;
    float* ffb    = w + 6*M1;             // [4096,1024]
    float* scores = w + 10*M1;            // [8,512,512]
    float* ench   = w + 12*M1;            // [256,128]
    float* dech   = ench + NPTS*128;
    float* rec    = dech + NPTS*128;      // [256,4096]
    float* part   = rec + M1;             // [1024]
    float* lowb   = part + 1024;          // [256,2]

    embed_kernel<<<dim3(ROWS), dim3(256), 0, stream>>>(x_seg, emb_W, emb_b, z);

    const float scale = 1.0f / sqrtf((float)DHD);

    for (int l = 0; l < LLAYERS; ++l) {
        const float* Wq_l = Wq + (size_t)l * DD * DD;
        const float* Wk_l = Wk + (size_t)l * DD * DD;
        const float* Wv_l = Wv + (size_t)l * DD * DD;
        const float* Wo_l = Wo + (size_t)l * DD * DD;
        const float* W1_l = W1 + (size_t)l * DD * FFN;
        const float* b1_l = b1 + (size_t)l * FFN;
        const float* W2_l = W2 + (size_t)l * FFN * DD;
        const float* b2_l = b2 + (size_t)l * DD;

        run_tdm(z, DD, 0, Wq_l, DD, 0, qb, DD, 0, nullptr, ROWS, DD, DD, 1.0f, 0, 1, stream);
        run_tdm(z, DD, 0, Wk_l, DD, 0, kb, DD, 0, nullptr, ROWS, DD, DD, 1.0f, 0, 1, stream);
        run_tdm(z, DD, 0, Wv_l, DD, 0, vb, DD, 0, nullptr, ROWS, DD, DD, 1.0f, 0, 1, stream);

        for (int b = 0; b < BB; ++b) {
            const float* qB = qb + (size_t)b * SEQ * DD;
            const float* kB = kb + (size_t)b * SEQ * DD;
            const float* vB = vb + (size_t)b * SEQ * DD;
            float*       oB = attb + (size_t)b * SEQ * DD;
            // scores[h] = scale * Q_bh @ K_bh^T   (batched over heads)
            run_tn(qB, DD, DHD, kB, DD, DHD,
                   scores, SEQ, (long)SEQ * SEQ, SEQ, SEQ, DHD, scale, HH, stream);
            softmax_kernel<<<dim3(SEQ, HH), dim3(256), 0, stream>>>(scores, SEQ);
            // O_bh = att @ V_bh
            run_tdm(scores, SEQ, (long)SEQ * SEQ, vB, DD, DHD,
                    oB, DD, DHD, nullptr, SEQ, DHD, SEQ, 1.0f, 0, HH, stream);
        }

        run_tdm(attb, DD, 0, Wo_l, DD, 0, ob, DD, 0, nullptr, ROWS, DD, DD, 1.0f, 0, 1, stream);
        add_ln_kernel<<<dim3(ROWS), dim3(256), 0, stream>>>(z, ob, ln1_g + l*DD, ln1_b + l*DD);

        run_tdm(z, DD, 0, W1_l, FFN, 0, ffb, FFN, 0, b1_l, ROWS, FFN, DD, 1.0f, 1, 1, stream);
        run_tdm(ffb, FFN, 0, W2_l, DD, 0, ob, DD, 0, b2_l, ROWS, DD, FFN, 1.0f, 0, 1, stream);
        add_ln_kernel<<<dim3(ROWS), dim3(256), 0, stream>>>(z, ob, ln2_g + l*DD, ln2_b + l*DD);
    }

    // autoencoder (pts_in == z viewed as [256,4096])
    run_tdm(z, ENC_IN, 0, enc_W1, 128, 0, ench, 128, 0, enc_b1,
            NPTS, 128, ENC_IN, 1.0f, 1, 1, stream);
    run_safe(ench, 128, enc_W2, 2, lowb, 2, enc_b2, NPTS, 2, 128, 2, stream);
    run_safe(lowb, 2, dec_W1, 128, dech, 128, dec_b1, NPTS, 128, 2, 1, stream);
    run_tdm(dech, 128, 0, dec_W2, ENC_IN, 0, rec, ENC_IN, 0, dec_b2,
            NPTS, ENC_IN, 128, 1.0f, 0, 1, stream);

    // losses
    l1_partial_kernel<<<dim3(1024), dim3(256), 0, stream>>>(z, rec, part, (long)NPTS * ENC_IN);
    finalize_kernel<<<dim3(1), dim3(256), 0, stream>>>(lowb, centers, part, 1024, out,
                                                       logf((float)NPTS),
                                                       (float)((long)NPTS * ENC_IN));
}
